// REGCN_38903813767427
// MI455X (gfx1250) — matrix operand order repaired and checked by reading, not compiled, not run on hardware
//
#include <hip/hip_runtime.h>
#include <hip/hip_bf16.h>

typedef __bf16 bf16_t;
typedef bf16_t v16bf __attribute__((ext_vector_type(16)));
typedef bf16_t v8bf  __attribute__((ext_vector_type(8)));
typedef float  v8f   __attribute__((ext_vector_type(8)));
typedef int    v4i_vs __attribute__((vector_size(16)));   // matches async-LDS builtin param

#define AS1 __attribute__((address_space(1)))
#define AS3 __attribute__((address_space(3)))

#define NN      50000
#define EE      500000
#define CDIM    256
#define SCALE_F 100.0f

#if defined(__has_builtin)
#if __has_builtin(__builtin_amdgcn_global_load_async_to_lds_b128)
#define HAVE_ASYNC_LDS 1
#endif
#endif

__device__ __forceinline__ float leaky01(float x) { return x > 0.0f ? x : 0.01f * x; }

// ---------------------------------------------------------------------------
// Prep: Wt[n][k] = bf16(W[k][n]) for W1 and W2 (each 256x256).
// ---------------------------------------------------------------------------
__global__ __launch_bounds__(256) void prep_wt_kernel(
    const float* __restrict__ W1, const float* __restrict__ W2,
    bf16_t* __restrict__ Wt1, bf16_t* __restrict__ Wt2)
{
    const int k = blockIdx.x & 255;
    const float* W  = (blockIdx.x < 256) ? W1  : W2;
    bf16_t*      Wt = (blockIdx.x < 256) ? Wt1 : Wt2;
    const int n = threadIdx.x;
    Wt[n * CDIM + k] = (bf16_t)W[k * CDIM + n];
}

// ---------------------------------------------------------------------------
// Gather node inputs (feature table vs embedding table), convert to bf16.
// ---------------------------------------------------------------------------
__global__ __launch_bounds__(256) void gather_kernel(
    const float* __restrict__ x0, const float* __restrict__ emb,
    const int* __restrict__ ntype, const int* __restrict__ lidx,
    bf16_t* __restrict__ hbf)
{
    const int node = blockIdx.x;
    const float* src = ((ntype[node] == 0) ? x0 : emb) + (size_t)lidx[node] * CDIM;
    hbf[(size_t)node * CDIM + threadIdx.x] = (bf16_t)src[threadIdx.x];
}

// ---------------------------------------------------------------------------
// Weighted in-degree: deg[c] += leaky_relu(rw[etype]*SCALE), both layers.
// ---------------------------------------------------------------------------
__global__ __launch_bounds__(256) void deg_kernel(
    const int* __restrict__ col, const int* __restrict__ etype,
    const float* __restrict__ rw1, const float* __restrict__ rw2,
    float* __restrict__ deg1, float* __restrict__ deg2)
{
    int e = blockIdx.x * blockDim.x + threadIdx.x;
    if (e >= EE) return;
    const int t = etype[e];
    const int c = col[e];
    unsafeAtomicAdd(&deg1[c], leaky01(rw1[t] * SCALE_F));
    unsafeAtomicAdd(&deg2[c], leaky01(rw2[t] * SCALE_F));
}

// ---------------------------------------------------------------------------
// Per-edge normalized weight: ewn[e] = w(e) / |deg[col(e)]|, both layers.
// ---------------------------------------------------------------------------
__global__ __launch_bounds__(256) void edge_norm_kernel(
    const int* __restrict__ col, const int* __restrict__ etype,
    const float* __restrict__ rw1, const float* __restrict__ rw2,
    const float* __restrict__ deg1, const float* __restrict__ deg2,
    float* __restrict__ ewn1, float* __restrict__ ewn2)
{
    int e = blockIdx.x * blockDim.x + threadIdx.x;
    if (e >= EE) return;
    const int t = etype[e];
    const int c = col[e];
    ewn1[e] = leaky01(rw1[t] * SCALE_F) / fabsf(deg1[c]);
    ewn2[e] = leaky01(rw2[t] * SCALE_F) / fabsf(deg2[c]);
}

// ---------------------------------------------------------------------------
// GEMM: C[M x 256] = A[M x 256](bf16) * W[256 x 256], W given transposed
// (Bt[n][k]).  80-row strip per block (5 M-tiles); 16 waves cover all 256
// output columns.  Each wave holds 5 accumulators and reuses every B
// fragment across 5 x v_wmma_f32_16x16x32_bf16 (40 WMMAs/wave).  A strip is
// staged into padded LDS via async-to-LDS loads when available.
// ---------------------------------------------------------------------------
#define LDS_PITCH 264   // 256 + 8 bf16 pad -> 528B row stride, avoids conflicts
#define MTILES    5     // 5 * 16 = 80 rows per block; 50000 / 80 = 625 exact

__global__ __launch_bounds__(512) void gemm_bf16_wmma_kernel(
    const bf16_t* __restrict__ A,   // [M][256] row-major bf16
    const bf16_t* __restrict__ Bt,  // [256][256] Bt[n][k] = W[k][n]
    float* __restrict__ C)          // [M][256] fp32
{
    __shared__ bf16_t As[MTILES * 16 * LDS_PITCH];   // 80 rows, ~42 KB

    const int mbase = blockIdx.x * (MTILES * 16);

    // Stage the 80x256 bf16 A strip: 2560 16B chunks over 512 threads (5 each).
#pragma unroll
    for (int i = 0; i < MTILES; ++i) {
        const int idx = threadIdx.x + i * 512;
        const int r   = idx >> 5;       // 0..79 row
        const int ch  = idx & 31;       // 0..31 16B chunk
        const bf16_t* gsrc = A + (size_t)(mbase + r) * CDIM + ch * 8;
        bf16_t*       ldst = As + r * LDS_PITCH + ch * 8;
#ifdef HAVE_ASYNC_LDS
        __builtin_amdgcn_global_load_async_to_lds_b128(
            (AS1 v4i_vs*)(AS1 void*)gsrc,
            (AS3 v4i_vs*)(AS3 void*)ldst, 0, 0);
#else
        *(uint4*)ldst = *(const uint4*)gsrc;
#endif
    }
#ifdef HAVE_ASYNC_LDS
#if __has_builtin(__builtin_amdgcn_s_wait_asynccnt)
    __builtin_amdgcn_s_wait_asynccnt(0);
#else
    asm volatile("s_wait_asynccnt 0x0" ::: "memory");
#endif
#endif
    __syncthreads();

    const int wave  = threadIdx.x >> 5;   // 0..15 -> N tile
    const int lane  = threadIdx.x & 31;
    const int l16   = lane & 15;
    const int lhalf = lane >> 4;          // 0 or 1
    const int nbase = wave * 16;

    v8f acc[MTILES];
#pragma unroll
    for (int mt = 0; mt < MTILES; ++mt) acc[mt] = (v8f){};

    // A fragment source: row = mt*16 + l16; lane-half picks K sub-offset {0,8}.
    const bf16_t* arow = As + l16 * LDS_PITCH + lhalf * 8;
    // B fragment source: col = nbase+l16; lane-half picks K sub-offset {0,8}.
    const bf16_t* brow = Bt + (size_t)(nbase + l16) * CDIM + lhalf * 8;

#pragma unroll
    for (int kb = 0; kb < CDIM; kb += 32) {
        // One B 32x16 fragment, reused for all 5 M-tiles.
        v8bf blo = *(const v8bf*)(brow + kb);
        v8bf bhi = *(const v8bf*)(brow + kb + 16);
        v16bf b = __builtin_shufflevector(blo, bhi,
                                          0, 1, 2, 3, 4, 5, 6, 7,
                                          8, 9, 10, 11, 12, 13, 14, 15);
#pragma unroll
        for (int mt = 0; mt < MTILES; ++mt) {
            const bf16_t* ap = arow + mt * 16 * LDS_PITCH + kb;
            v8bf alo = *(const v8bf*)(ap);
            v8bf ahi = *(const v8bf*)(ap + 16);
            v16bf a = __builtin_shufflevector(alo, ahi,
                                              0, 1, 2, 3, 4, 5, 6, 7,
                                              8, 9, 10, 11, 12, 13, 14, 15);
            acc[mt] = __builtin_amdgcn_wmma_f32_16x16x32_bf16(
                /*neg_a=*/false, a, /*neg_b=*/false, b,
                /*c_mod=*/(short)0, acc[mt], /*reuse_a=*/false, /*reuse_b=*/false);
        }
    }

    // D layout: VGPR r -> row mbase + mt*16 + lhalf*8 + r, col nbase + l16.
#pragma unroll
    for (int mt = 0; mt < MTILES; ++mt) {
        float* cout = C + (size_t)(mbase + mt * 16 + lhalf * 8) * CDIM + nbase + l16;
#pragma unroll
        for (int r = 0; r < 8; ++r)
            cout[(size_t)r * CDIM] = acc[mt][r];
    }
}

// ---------------------------------------------------------------------------
// Scatter-add:  agg[col(e)][:] += ewn[e] * xsrc[row(e)][:]
// One 256-thread block per edge iteration (coalesced 1KB gather + 1KB atomic
// burst); prefetch the next edge's source row one iteration ahead.
// ---------------------------------------------------------------------------
__global__ __launch_bounds__(256) void scatter_kernel(
    const float* __restrict__ xsrc,
    const int* __restrict__ row, const int* __restrict__ col,
    const float* __restrict__ ewn,
    float* __restrict__ agg)
{
    const int t = threadIdx.x;
    for (int e = blockIdx.x; e < EE; e += gridDim.x) {
        const int en = e + gridDim.x;
        if (en < EE)
            __builtin_prefetch(&xsrc[(size_t)row[en] * CDIM + t], 0, 0);
        const int r = row[e];
        const int c = col[e];
        const float w = ewn[e];
        const float m = w * xsrc[(size_t)r * CDIM + t];
        unsafeAtomicAdd(&agg[(size_t)c * CDIM + t], m);
    }
}

// ---------------------------------------------------------------------------
// h2 = bf16(relu(agg + bias))  (between layers)
// ---------------------------------------------------------------------------
__global__ __launch_bounds__(256) void bias_relu_bf16_kernel(
    const float* __restrict__ agg, const float* __restrict__ bias,
    bf16_t* __restrict__ h2)
{
    const size_t total = (size_t)NN * CDIM;
    for (size_t i = (size_t)blockIdx.x * blockDim.x + threadIdx.x; i < total;
         i += (size_t)gridDim.x * blockDim.x) {
        float v = agg[i] + bias[i & (CDIM - 1)];
        h2[i] = (bf16_t)(v > 0.0f ? v : 0.0f);
    }
}

// ---------------------------------------------------------------------------
// out = log_softmax(agg + bias) per row. One block per row, LDS reductions.
// ---------------------------------------------------------------------------
__global__ __launch_bounds__(256) void logsoftmax_kernel(
    const float* __restrict__ agg, const float* __restrict__ bias,
    float* __restrict__ out)
{
    __shared__ float red[CDIM];
    const int rowid = blockIdx.x;
    const int t = threadIdx.x;
    const float v = agg[(size_t)rowid * CDIM + t] + bias[t];

    red[t] = v;
    __syncthreads();
#pragma unroll
    for (int s = 128; s > 0; s >>= 1) {
        if (t < s) red[t] = fmaxf(red[t], red[t + s]);
        __syncthreads();
    }
    const float m = red[0];
    __syncthreads();

    red[t] = __expf(v - m);
    __syncthreads();
#pragma unroll
    for (int s = 128; s > 0; s >>= 1) {
        if (t < s) red[t] += red[t + s];
        __syncthreads();
    }
    out[(size_t)rowid * CDIM + t] = v - m - __logf(red[0]);
}

// ---------------------------------------------------------------------------
// Host launcher
// ---------------------------------------------------------------------------
extern "C" void kernel_launch(void* const* d_in, const int* in_sizes, int n_in,
                              void* d_out, int out_size, void* d_ws, size_t ws_size,
                              hipStream_t stream)
{
    (void)in_sizes; (void)n_in; (void)out_size; (void)ws_size;

    const float* x0   = (const float*)d_in[0];
    const float* emb1 = (const float*)d_in[1];
    const float* W1   = (const float*)d_in[2];
    /* Wr1 = d_in[3] dead */
    const float* b1   = (const float*)d_in[4];
    const float* rw1  = (const float*)d_in[5];
    const float* W2   = (const float*)d_in[6];
    /* Wr2 = d_in[7] dead */
    const float* b2   = (const float*)d_in[8];
    const float* rw2  = (const float*)d_in[9];
    const int* ntype  = (const int*)d_in[10];
    const int* lidx   = (const int*)d_in[11];
    const int* eidx   = (const int*)d_in[12];   // [2][E]
    const int* etype  = (const int*)d_in[13];
    float* out = (float*)d_out;

    const int* erow = eidx;        // edge_index[0]
    const int* ecol = eidx + EE;   // edge_index[1]

    // Workspace carve-up (256B aligned slabs).
    char* ws = (char*)d_ws;
    size_t off = 0;
    auto carve = [&](size_t bytes) -> char* {
        char* p = ws + off;
        off = (off + bytes + 255) & ~(size_t)255;
        return p;
    };
    bf16_t* hbf  = (bf16_t*)carve((size_t)NN * CDIM * sizeof(bf16_t));  // 25.6MB (reused as h2)
    float*  xsrc = (float*)carve((size_t)NN * CDIM * sizeof(float));    // 51.2MB
    float*  agg  = (float*)carve((size_t)NN * CDIM * sizeof(float));    // 51.2MB
    bf16_t* Wt1  = (bf16_t*)carve((size_t)CDIM * CDIM * sizeof(bf16_t));
    bf16_t* Wt2  = (bf16_t*)carve((size_t)CDIM * CDIM * sizeof(bf16_t));
    float*  deg1 = (float*)carve((size_t)NN * sizeof(float));
    float*  deg2 = (float*)carve((size_t)NN * sizeof(float));
    float*  ewn1 = (float*)carve((size_t)EE * sizeof(float));
    float*  ewn2 = (float*)carve((size_t)EE * sizeof(float));

    // Zero accumulators (graph-capture-safe async memsets).
    (void)hipMemsetAsync(deg1, 0, (size_t)NN * sizeof(float), stream);
    (void)hipMemsetAsync(deg2, 0, (size_t)NN * sizeof(float), stream);
    (void)hipMemsetAsync(agg, 0, (size_t)NN * CDIM * sizeof(float), stream);

    // Prep: weight transpose+convert, input gather.
    prep_wt_kernel<<<512, 256, 0, stream>>>(W1, W2, Wt1, Wt2);
    gather_kernel<<<NN, 256, 0, stream>>>(x0, emb1, ntype, lidx, hbf);

    // Edge weights / degree normalization (both layers at once).
    const int eblocks = (EE + 255) / 256;
    deg_kernel<<<eblocks, 256, 0, stream>>>(ecol, etype, rw1, rw2, deg1, deg2);
    edge_norm_kernel<<<eblocks, 256, 0, stream>>>(ecol, etype, rw1, rw2,
                                                  deg1, deg2, ewn1, ewn2);

    const int gemm_blocks = NN / (MTILES * 16);  // 625, exact

    // ----- Layer 1 -----
    gemm_bf16_wmma_kernel<<<gemm_blocks, 512, 0, stream>>>(hbf, Wt1, xsrc);
    scatter_kernel<<<8192, 256, 0, stream>>>(xsrc, erow, ecol, ewn1, agg);
    bias_relu_bf16_kernel<<<4096, 256, 0, stream>>>(agg, b1, hbf);  // hbf reused as h2
    (void)hipMemsetAsync(agg, 0, (size_t)NN * CDIM * sizeof(float), stream);

    // ----- Layer 2 -----
    gemm_bf16_wmma_kernel<<<gemm_blocks, 512, 0, stream>>>(hbf, Wt2, xsrc);
    scatter_kernel<<<8192, 256, 0, stream>>>(xsrc, erow, ecol, ewn2, agg);

    // ----- log_softmax -----
    logsoftmax_kernel<<<NN, 256, 0, stream>>>(agg, b2, out);
}
